// H0_19911468384642
// MI455X (gfx1250) — compile-verified, matrix operand order checked by reading, not run on hardware
//
#include <hip/hip_runtime.h>
#include <cmath>

// Problem constants (match reference)
static constexpr int Bn = 8;
static constexpr int Tn = 8192;
static constexpr int Ln = 4;
static constexpr int Kn = 64;
static constexpr int Fn = 2;
static constexpr int Sc = 64;          // chunk length
static constexpr int Cc = Tn / Sc;     // 128 chunks
#define LOG2PI_F 1.83787706640934548356f
#define NEG_BIG  -1.0e30f

typedef __attribute__((ext_vector_type(16))) _Float16 v16h;
typedef __attribute__((ext_vector_type(8)))  float    v8f;
typedef __attribute__((ext_vector_type(4)))  int      v4i;

// Async global<->LDS (gfx1250). Builtin signature (from probe):
//   (v4i addrspace(1)*, v4i addrspace(3)*, imm int offset, imm int cpol)
#if __has_builtin(__builtin_amdgcn_global_load_async_to_lds_b128)
#define ASYNC_LOAD_LDS 1
#endif
#if __has_builtin(__builtin_amdgcn_global_store_async_from_lds_b128)
#define ASYNC_STORE_LDS 1
#endif
typedef __attribute__((address_space(1))) v4i v4i_g;
typedef __attribute__((address_space(3))) v4i v4i_s;
#define WAIT_ASYNC0() asm volatile("s_wait_asynccnt 0x0" ::: "memory")
#define WAIT_ASYNC1() asm volatile("s_wait_asynccnt 0x1" ::: "memory")

// K-index mapping for 16-bit WMMA A/B operands (CDNA5 ISA 7.12.2):
// half-index h (0..15) within v16h, lane-half hi (0/1).
__device__ __forceinline__ int kmap(int h, int hi) {
  return ((h >> 3) << 4) + (h & 7) + (hi << 3);
}

// Load per-layer constants into LDS: Wlds[i][j] = softmax(trans[l][i])[j],
// emission mu / exp(-log_std) / (sum log_std + log 2pi).
__device__ __forceinline__ void load_layer_params(
    int l, int tid, const float* __restrict__ trans,
    const float* __restrict__ mu, const float* __restrict__ lstd,
    _Float16 (*Wlds)[Kn], float (*muS)[Fn], float (*isS)[Fn], float* cS) {
  if (tid < Kn) {
    float c = LOG2PI_F;
    for (int f = 0; f < Fn; ++f) {
      float ls = lstd[((size_t)l * Kn + tid) * Fn + f];
      muS[tid][f] = mu[((size_t)l * Kn + tid) * Fn + f];
      isS[tid][f] = __expf(-ls);
      c += ls;
    }
    cS[tid] = c;
    const float* row = trans + ((size_t)l * Kn + tid) * Kn;
    float mx = -3.4e38f;
    for (int j = 0; j < Kn; ++j) mx = fmaxf(mx, row[j]);
    float sm = 0.0f;
    for (int j = 0; j < Kn; ++j) sm += __expf(row[j] - mx);
    float lse = mx + __logf(sm);
    for (int j = 0; j < Kn; ++j) Wlds[tid][j] = (_Float16)__expf(row[j] - lse);
  }
}

__device__ __forceinline__ float emit_lb(float x0, float x1, int m,
                                         const float (*muS)[Fn],
                                         const float (*isS)[Fn],
                                         const float* cS) {
  float z0 = (x0 - muS[m][0]) * isS[m][0];
  float z1 = (x1 - muS[m][1]) * isS[m][1];
  return -0.5f * (z0 * z0 + z1 * z1) - cS[m];
}

// ---------------------------------------------------------------------------
// Kernel 0: per-layer EMA -> standardized features xf[B][T][L][F]
// ---------------------------------------------------------------------------
__global__ void ema_feat_kernel(const float* __restrict__ lv,
                                const float* __restrict__ lvi,
                                float* __restrict__ xf) {
  const int b = blockIdx.x;
  const int lane = threadIdx.x;
  if (lane >= Ln * Fn) return;
  const int l = lane & (Ln - 1);
  const int f = lane >> 2;               // 0 = log_volume, 1 = imbalance
  const float* __restrict__ src = (f == 0) ? lv : lvi;
  const float a = 1.0f - exp2f(-(float)(2 + l));
  const float one_m_a = 1.0f - a;
  float m = 0.0f, s = 0.0f;
  for (int t = 0; t < Tn; ++t) {
    if (t + 64 < Tn) __builtin_prefetch(&src[b * Tn + t + 64], 0, 3);
    float x = src[b * Tn + t];
    m = a * m + one_m_a * x;
    s = a * s + one_m_a * x * x;
    float istd = rsqrtf(fmaxf(s - m * m, 1e-6f));
    float feat = (f == 0) ? (x - m) * istd : x * istd;
    xf[((size_t)(b * Tn + t) * Ln + l) * Fn + f] = feat;
  }
}

// ---------------------------------------------------------------------------
// Phase A: chunk operators. grid = (Cc, Bn, Ln), 128 threads (4 waves).
// Composes logG_c = log( prod_{t in chunk} D_t A^T ) as a 64x64 log-matrix,
// one 64x64x64 f16 GEMM per step (4 N-tiles x 2 K-halves = 8 WMMA per wave).
// Chunk 0 starts at t=1 (t=0 uses init distribution, handled in Phase B).
// ---------------------------------------------------------------------------
__global__ __launch_bounds__(128)
void chunk_op_kernel(const float* __restrict__ xf,
                     const float* __restrict__ trans,
                     const float* __restrict__ mu,
                     const float* __restrict__ lstd,
                     float* __restrict__ logG) {
  __shared__ _Float16 Wlds[Kn][Kn];
  __shared__ float Mlds[Kn][Kn];     // current log-operator
  __shared__ float cmA[Kn];          // per-column max
  __shared__ float pm[8][Kn];
  __shared__ float muS[Kn][Fn], isS[Kn][Fn], cS[Kn];

  const int c = blockIdx.x, b = blockIdx.y, l = blockIdx.z;
  const int tid = threadIdx.x;
  const int w = tid >> 5, lane = tid & 31;
  const int hi = lane >> 4, n = lane & 15;
  const int seg = (w << 1) + hi;

  load_layer_params(l, tid, trans, mu, lstd, Wlds, muS, isS, cS);
  __syncthreads();

  // A-operand, forward orientation: A[m][i] = softA[i][m]
  const int aM = (w << 4) + n;
  v16h a0, a1;
#pragma unroll
  for (int h = 0; h < 16; ++h) {
    int k = kmap(h, hi);
    a0[h] = Wlds[k][aM];
    a1[h] = Wlds[k + 32][aM];
  }
  int mIdx[8];
#pragma unroll
  for (int r = 0; r < 8; ++r) mIdx[r] = (w << 4) + (hi << 3) + r;

  // init to log(identity)
#pragma unroll
  for (int nt = 0; nt < 4; ++nt) {
    int col = (nt << 4) + n;
#pragma unroll
    for (int r = 0; r < 8; ++r)
      Mlds[mIdx[r]][col] = (mIdx[r] == col) ? 0.0f : NEG_BIG;
  }
  if (tid < Kn) cmA[tid] = 0.0f;
  __syncthreads();

  const int tBeg = (c == 0) ? 1 : (c * Sc);
  const int tEnd = (c + 1) * Sc;
  for (int t = tBeg; t < tEnd; ++t) {
    const size_t xoff = ((size_t)(b * Tn + t) * Ln + l) * Fn;
    float x0 = xf[xoff + 0];
    float x1 = xf[xoff + 1];
    float lbv[8];
#pragma unroll
    for (int r = 0; r < 8; ++r) lbv[r] = emit_lb(x0, x1, mIdx[r], muS, isS, cS);

    float val[4][8];
#pragma unroll
    for (int nt = 0; nt < 4; ++nt) {
      v16h b0, b1;
#pragma unroll
      for (int h = 0; h < 16; ++h) {
        int k = kmap(h, hi);
        int col = (nt << 4) + n;
        b0[h] = (_Float16)__expf(Mlds[k][col] - cmA[col]);
        b1[h] = (_Float16)__expf(Mlds[k + 32][col] - cmA[col]);
      }
      v8f cc = {};
      cc = __builtin_amdgcn_wmma_f32_16x16x32_f16(false, a0, false, b0,
                                                  (short)0, cc, false, false);
      cc = __builtin_amdgcn_wmma_f32_16x16x32_f16(false, a1, false, b1,
                                                  (short)0, cc, false, false);
      int col = (nt << 4) + n;
#pragma unroll
      for (int r = 0; r < 8; ++r)
        val[nt][r] = __logf(cc[r] + 1e-30f) + cmA[col] + lbv[r];
    }
    __syncthreads();                 // all reads of Mlds/cmA complete
#pragma unroll
    for (int nt = 0; nt < 4; ++nt) {
      int col = (nt << 4) + n;
      float mx = val[nt][0];
#pragma unroll
      for (int r = 1; r < 8; ++r) mx = fmaxf(mx, val[nt][r]);
      pm[seg][col] = mx;
#pragma unroll
      for (int r = 0; r < 8; ++r) Mlds[mIdx[r]][col] = val[nt][r];
    }
    __syncthreads();
    if (tid < Kn) {
      float m2 = pm[0][tid];
      for (int s2 = 1; s2 < 8; ++s2) m2 = fmaxf(m2, pm[s2][tid]);
      cmA[tid] = m2;
    }
    __syncthreads();
  }

  const size_t base = ((size_t)(l * Bn + b) * Cc + c) * Kn * Kn;
#pragma unroll
  for (int nt = 0; nt < 4; ++nt) {
    int col = (nt << 4) + n;
#pragma unroll
    for (int r = 0; r < 8; ++r)
      logG[base + (size_t)mIdx[r] * Kn + col] = Mlds[mIdx[r]][col];
  }
}

// ---------------------------------------------------------------------------
// Phase B (forward): chunk-boundary alphas. grid = Ln*Bn, 64 threads.
// ---------------------------------------------------------------------------
__global__ __launch_bounds__(64)
void compose_fwd_kernel(const float* __restrict__ xf,
                        const float* __restrict__ initl,
                        const float* __restrict__ mu,
                        const float* __restrict__ lstd,
                        const float* __restrict__ logG,
                        float* __restrict__ wsAin) {
  __shared__ float as_[Kn];
  __shared__ float piLSE;
  const int l = blockIdx.x >> 3;
  const int b = blockIdx.x & (Bn - 1);
  const int m = threadIdx.x;

  if (m == 0) {
    const float* row = initl + (size_t)l * Kn;
    float mx = -3.4e38f;
    for (int j = 0; j < Kn; ++j) mx = fmaxf(mx, row[j]);
    float sm = 0.0f;
    for (int j = 0; j < Kn; ++j) sm += __expf(row[j] - mx);
    piLSE = mx + __logf(sm);
  }
  float ls0 = lstd[((size_t)l * Kn + m) * Fn + 0];
  float ls1 = lstd[((size_t)l * Kn + m) * Fn + 1];
  float m0 = mu[((size_t)l * Kn + m) * Fn + 0];
  float m1 = mu[((size_t)l * Kn + m) * Fn + 1];
  float x0 = xf[((size_t)(b * Tn + 0) * Ln + l) * Fn + 0];
  float x1 = xf[((size_t)(b * Tn + 0) * Ln + l) * Fn + 1];
  float z0 = (x0 - m0) * __expf(-ls0);
  float z1 = (x1 - m1) * __expf(-ls1);
  float lb = -0.5f * (z0 * z0 + z1 * z1) - (ls0 + ls1 + LOG2PI_F);
  __syncthreads();
  float v = initl[(size_t)l * Kn + m] - piLSE + lb;
  as_[m] = v;
  wsAin[(((size_t)l * Cc + 0) * Bn + b) * Kn + m] = v;
  __syncthreads();

  for (int c = 0; c < Cc - 1; ++c) {
    const float* row = logG + (((size_t)(l * Bn + b) * Cc + c) * Kn + m) * Kn;
    float mx = NEG_BIG;
    for (int i = 0; i < Kn; ++i) mx = fmaxf(mx, row[i] + as_[i]);
    float sm = 0.0f;
    for (int i = 0; i < Kn; ++i) sm += __expf(row[i] + as_[i] - mx);
    float nv = mx + __logf(sm);
    __syncthreads();
    as_[m] = nv;
    wsAin[(((size_t)l * Cc + (c + 1)) * Bn + b) * Kn + m] = nv;
    __syncthreads();
  }
}

// ---------------------------------------------------------------------------
// Phase B (backward): chunk-end betas via transposed operators (G_c^T).
// ---------------------------------------------------------------------------
__global__ __launch_bounds__(64)
void compose_bwd_kernel(const float* __restrict__ logG,
                        float* __restrict__ wsBin) {
  __shared__ float bs_[Kn];
  const int l = blockIdx.x >> 3;
  const int b = blockIdx.x & (Bn - 1);
  const int m = threadIdx.x;
  bs_[m] = 0.0f;
  wsBin[(((size_t)l * Cc + (Cc - 1)) * Bn + b) * Kn + m] = 0.0f;
  __syncthreads();
  for (int c = Cc - 1; c >= 1; --c) {
    const float* gc = logG + ((size_t)(l * Bn + b) * Cc + c) * Kn * Kn;
    float mx = NEG_BIG;
    for (int j = 0; j < Kn; ++j) mx = fmaxf(mx, gc[(size_t)j * Kn + m] + bs_[j]);
    float sm = 0.0f;
    for (int j = 0; j < Kn; ++j)
      sm += __expf(gc[(size_t)j * Kn + m] + bs_[j] - mx);
    float nv = mx + __logf(sm);
    __syncthreads();
    bs_[m] = nv;
    wsBin[(((size_t)l * Cc + (c - 1)) * Bn + b) * Kn + m] = nv;
    __syncthreads();
  }
}

// ---------------------------------------------------------------------------
// Phase C (forward replay): grid = (Cc, Ln), 128 threads, batches in N dim.
// Alpha checkpoints leave via async store-from-LDS (2-deep staging ring) so
// the 2KB/step write happens off the scan's critical chain.
// ---------------------------------------------------------------------------
__global__ __launch_bounds__(128)
void replay_fwd_kernel(const float* __restrict__ xf,
                       const float* __restrict__ trans,
                       const float* __restrict__ mu,
                       const float* __restrict__ lstd,
                       const float* __restrict__ wsAin,
                       float* __restrict__ wsA,
                       float* __restrict__ wsZ) {
  __shared__ _Float16 Wlds[Kn][Kn];
  __shared__ float As[Kn][16];
  __shared__ float cm[16];
  __shared__ float pm[8][16];
  __shared__ float muS[Kn][Fn], isS[Kn][Fn], cS[Kn];
#if ASYNC_STORE_LDS
  __shared__ __align__(16) float sTile[2][Kn * Bn];  // staging ring
#endif

  const int c = blockIdx.x, l = blockIdx.y;
  const int tid = threadIdx.x;
  const int w = tid >> 5, lane = tid & 31;
  const int hi = lane >> 4, n = lane & 15;
  const int bidx = n & (Bn - 1);
  const int seg = (w << 1) + hi;

  load_layer_params(l, tid, trans, mu, lstd, Wlds, muS, isS, cS);
  __syncthreads();

  const int aM = (w << 4) + n;
  v16h a0, a1;
#pragma unroll
  for (int h = 0; h < 16; ++h) {
    int k = kmap(h, hi);
    a0[h] = Wlds[k][aM];
    a1[h] = Wlds[k + 32][aM];
  }
  int mIdx[8];
#pragma unroll
  for (int r = 0; r < 8; ++r) mIdx[r] = (w << 4) + (hi << 3) + r;

  // load chunk-entry alpha
  float v0[8];
#pragma unroll
  for (int r = 0; r < 8; ++r)
    v0[r] = wsAin[(((size_t)l * Cc + c) * Bn + bidx) * Kn + mIdx[r]];
  float mx = v0[0];
#pragma unroll
  for (int r = 1; r < 8; ++r) mx = fmaxf(mx, v0[r]);
  pm[seg][n] = mx;
#pragma unroll
  for (int r = 0; r < 8; ++r) As[mIdx[r]][n] = v0[r];
  __syncthreads();
  if (tid < 16) {
    float m2 = pm[0][tid];
    for (int s2 = 1; s2 < 8; ++s2) m2 = fmaxf(m2, pm[s2][tid]);
    cm[tid] = m2;
  }
  __syncthreads();

  int t0 = c * Sc;
  const int t1 = (c + 1) * Sc;
  if (c == 0) {                        // alpha_in[0] IS alpha_0: store t=0
    if (n < Bn) {
      const size_t base = ((size_t)l * Tn + 0) * Kn;
#pragma unroll
      for (int r = 0; r < 8; ++r) wsA[(base + mIdx[r]) * Bn + n] = v0[r];
    }
    t0 = 1;
  }

  for (int t = t0; t < t1; ++t) {
    const size_t xoff = ((size_t)(bidx * Tn + t) * Ln + l) * Fn;
    float x0 = xf[xoff + 0];
    float x1 = xf[xoff + 1];
    if (t + 8 < t1) __builtin_prefetch(&xf[xoff + (size_t)8 * Ln * Fn], 0, 3);

    v16h b0, b1;
#pragma unroll
    for (int h = 0; h < 16; ++h) {
      int k = kmap(h, hi);
      b0[h] = (_Float16)__expf(As[k][n] - cm[n]);
      b1[h] = (_Float16)__expf(As[k + 32][n] - cm[n]);
    }
    v8f cc = {};
    cc = __builtin_amdgcn_wmma_f32_16x16x32_f16(false, a0, false, b0,
                                                (short)0, cc, false, false);
    cc = __builtin_amdgcn_wmma_f32_16x16x32_f16(false, a1, false, b1,
                                                (short)0, cc, false, false);
    float val[8];
#pragma unroll
    for (int r = 0; r < 8; ++r)
      val[r] = __logf(cc[r] + 1e-30f) + cm[n] + emit_lb(x0, x1, mIdx[r], muS, isS, cS);

    mx = val[0];
#pragma unroll
    for (int r = 1; r < 8; ++r) mx = fmaxf(mx, val[r]);
    pm[seg][n] = mx;
#if ASYNC_STORE_LDS
    WAIT_ASYNC1();                     // ring slot (t&1) store from t-2 done
#endif
    __syncthreads();
#pragma unroll
    for (int r = 0; r < 8; ++r) As[mIdx[r]][n] = val[r];
#if ASYNC_STORE_LDS
    if (n < Bn) {
#pragma unroll
      for (int r = 0; r < 8; ++r) sTile[t & 1][mIdx[r] * Bn + n] = val[r];
    }
#endif
    if (tid < 16) {
      float m2 = pm[0][tid];
      for (int s2 = 1; s2 < 8; ++s2) m2 = fmaxf(m2, pm[s2][tid]);
      cm[tid] = m2;
    }
    __syncthreads();

#if ASYNC_STORE_LDS
    {  // one b128 per lane: 128 threads x 16B = 2KB checkpoint tile
      float* gp = wsA + ((size_t)l * Tn + t) * Kn * Bn + (size_t)tid * 4;
      __builtin_amdgcn_global_store_async_from_lds_b128(
          (v4i_g*)gp, (v4i_s*)&sTile[t & 1][tid * 4], 0, 0);
    }
#else
    if (n < Bn) {
      const size_t base = ((size_t)l * Tn + t) * Kn;
#pragma unroll
      for (int r = 0; r < 8; ++r) wsA[(base + mIdx[r]) * Bn + n] = val[r];
    }
#endif
  }

  if (c == Cc - 1 && tid < Bn) {       // logZ[b] = LSE_m(alpha_{T-1})
    float s = 0.0f;
    for (int m = 0; m < Kn; ++m) s += __expf(As[m][tid] - cm[tid]);
    wsZ[l * Bn + tid] = cm[tid] + __logf(s);
  }
}

// ---------------------------------------------------------------------------
// Phase C (backward replay + gamma): grid = (Cc, Ln), 128 threads.
// Alpha checkpoint tiles arrive via double-buffered async load-to-LDS,
// prefetched one step ahead of the beta recursion.
// ---------------------------------------------------------------------------
__global__ __launch_bounds__(128)
void replay_bwd_kernel(const float* __restrict__ xf,
                       const float* __restrict__ trans,
                       const float* __restrict__ mu,
                       const float* __restrict__ lstd,
                       const float* __restrict__ wsA,
                       const float* __restrict__ wsBin,
                       const float* __restrict__ wsZ,
                       float* __restrict__ out_paths,
                       float* __restrict__ out_loss) {
  __shared__ _Float16 Wlds[Kn][Kn];
  __shared__ float As[Kn][16];
  __shared__ float cm[16];
  __shared__ float pm[8][16];
  __shared__ float ps[8][16];
  __shared__ float gm[16], gs[16];
  __shared__ float muS[Kn][Fn], isS[Kn][Fn], cS[Kn];
#if ASYNC_LOAD_LDS
  __shared__ __align__(16) float aTile[2][Kn * Bn];  // alpha double buffer
#endif

  const int c = blockIdx.x, l = blockIdx.y;
  const int tid = threadIdx.x;
  const int w = tid >> 5, lane = tid & 31;
  const int hi = lane >> 4, n = lane & 15;
  const int bidx = n & (Bn - 1);
  const int seg = (w << 1) + hi;

#if ASYNC_LOAD_LDS
  {  // prefetch alpha tile for t = (c+1)*Sc-1, overlapping the prologue
    float* gp = const_cast<float*>(wsA) +
                ((size_t)l * Tn + ((c + 1) * Sc - 1)) * Kn * Bn +
                (size_t)tid * 4;
    __builtin_amdgcn_global_load_async_to_lds_b128(
        (v4i_g*)gp, (v4i_s*)&aTile[0][tid * 4], 0, 0);
  }
#endif

  load_layer_params(l, tid, trans, mu, lstd, Wlds, muS, isS, cS);
  __syncthreads();

  // backward orientation: A[i][j] = softA[i][j]
  const int aM = (w << 4) + n;
  v16h a0, a1;
#pragma unroll
  for (int h = 0; h < 16; ++h) {
    int k = kmap(h, hi);
    a0[h] = Wlds[aM][k];
    a1[h] = Wlds[aM][k + 32];
  }
  int mIdx[8];
#pragma unroll
  for (int r = 0; r < 8; ++r) mIdx[r] = (w << 4) + (hi << 3) + r;

#pragma unroll
  for (int r = 0; r < 8; ++r)
    As[mIdx[r]][n] = wsBin[(((size_t)l * Cc + c) * Bn + bidx) * Kn + mIdx[r]];
  __syncthreads();

  int buf = 0;
  for (int t = (c + 1) * Sc - 1;; --t) {
#if ASYNC_LOAD_LDS
    WAIT_ASYNC0();                     // this wave's copy complete
    __syncthreads();                   // all waves' copies visible
    if (t > c * Sc) {                  // prefetch next tile (t-1)
      float* gp = const_cast<float*>(wsA) +
                  ((size_t)l * Tn + (t - 1)) * Kn * Bn + (size_t)tid * 4;
      __builtin_amdgcn_global_load_async_to_lds_b128(
          (v4i_g*)gp, (v4i_s*)&aTile[buf ^ 1][tid * 4], 0, 0);
    }
#endif
    const size_t xoff = ((size_t)(bidx * Tn + t) * Ln + l) * Fn;
    float x0 = xf[xoff + 0];
    float x1 = xf[xoff + 1];

    float g[8], lbv[8];
#pragma unroll
    for (int r = 0; r < 8; ++r) {
      int m = mIdx[r];
      lbv[r] = emit_lb(x0, x1, m, muS, isS, cS);
#if ASYNC_LOAD_LDS
      g[r] = aTile[buf][m * Bn + bidx] + As[m][n];
#else
      size_t aoff = (((size_t)l * Tn + t) * Kn + m) * Bn + bidx;
      if (t - 8 >= c * Sc)
        __builtin_prefetch(&wsA[aoff - (size_t)8 * Kn * Bn], 0, 3);
      g[r] = wsA[aoff] + As[m][n];     // log-alpha + log-beta
#endif
    }

    // gamma softmax over states
    float mx = g[0];
#pragma unroll
    for (int r = 1; r < 8; ++r) mx = fmaxf(mx, g[r]);
    pm[seg][n] = mx;
    __syncthreads();
    if (tid < 16) {
      float m2 = pm[0][tid];
      for (int s2 = 1; s2 < 8; ++s2) m2 = fmaxf(m2, pm[s2][tid]);
      gm[tid] = m2;
    }
    __syncthreads();
    float sm = 0.0f;
#pragma unroll
    for (int r = 0; r < 8; ++r) sm += __expf(g[r] - gm[n]);
    ps[seg][n] = sm;
    __syncthreads();
    if (tid < 16) {
      float s2 = 0.0f;
      for (int q = 0; q < 8; ++q) s2 += ps[q][tid];
      gs[tid] = s2;
    }
    __syncthreads();
    if (n < Bn) {                      // soft_paths[b][t][k][l]
#pragma unroll
      for (int r = 0; r < 8; ++r)
        out_paths[(((size_t)n * Tn + t) * Kn + mIdx[r]) * Ln + l] =
            __expf(g[r] - gm[n]) / gs[n];
    }

    if (t == c * Sc) break;

    // beta recursion
    float ev[8];
#pragma unroll
    for (int r = 0; r < 8; ++r) ev[r] = lbv[r] + As[mIdx[r]][n];
    mx = ev[0];
#pragma unroll
    for (int r = 1; r < 8; ++r) mx = fmaxf(mx, ev[r]);
    pm[seg][n] = mx;
    __syncthreads();
#pragma unroll
    for (int r = 0; r < 8; ++r) As[mIdx[r]][n] = ev[r];
    if (tid < 16) {
      float m2 = pm[0][tid];
      for (int s2 = 1; s2 < 8; ++s2) m2 = fmaxf(m2, pm[s2][tid]);
      cm[tid] = m2;
    }
    __syncthreads();

    v16h b0, b1;
#pragma unroll
    for (int h = 0; h < 16; ++h) {
      int k = kmap(h, hi);
      b0[h] = (_Float16)__expf(As[k][n] - cm[n]);
      b1[h] = (_Float16)__expf(As[k + 32][n] - cm[n]);
    }
    v8f cc = {};
    cc = __builtin_amdgcn_wmma_f32_16x16x32_f16(false, a0, false, b0,
                                                (short)0, cc, false, false);
    cc = __builtin_amdgcn_wmma_f32_16x16x32_f16(false, a1, false, b1,
                                                (short)0, cc, false, false);
    float val[8];
#pragma unroll
    for (int r = 0; r < 8; ++r) val[r] = __logf(cc[r] + 1e-30f) + cm[n];
    __syncthreads();                   // all B-operand reads done
#pragma unroll
    for (int r = 0; r < 8; ++r) As[mIdx[r]][n] = val[r];
    __syncthreads();
    buf ^= 1;
  }

  if (c == 0 && tid == 0) {            // hmm_losses[l] = -mean_b(logZ)/T
    float s = 0.0f;
    for (int b2 = 0; b2 < Bn; ++b2) s += wsZ[l * Bn + b2];
    out_loss[l] = -(s / (float)Bn) / (float)Tn;
  }
}

// ---------------------------------------------------------------------------
extern "C" void kernel_launch(void* const* d_in, const int* in_sizes, int n_in,
                              void* d_out, int out_size, void* d_ws,
                              size_t ws_size, hipStream_t stream) {
  const float* lv  = (const float*)d_in[0];  // log_volume [B,T]
  const float* lvi = (const float*)d_in[1];  // log_volume_imbalance [B,T]
  const float* tr  = (const float*)d_in[2];  // trans_logits [L,K,K]
  const float* il  = (const float*)d_in[3];  // init_logits [L,K]
  const float* mu  = (const float*)d_in[4];  // emit_mu [L,K,F]
  const float* ls  = (const float*)d_in[5];  // emit_log_std [L,K,F]

  float* ws   = (float*)d_ws;
  float* xf   = ws;                                      // B*T*L*F      (2 MB)
  float* logG = xf   + (size_t)Bn * Tn * Ln * Fn;        // L*B*C*K*K   (64 MB)
  float* ain  = logG + (size_t)Ln * Bn * Cc * Kn * Kn;   // L*C*B*K      (1 MB)
  float* bin  = ain  + (size_t)Ln * Cc * Bn * Kn;        // L*C*B*K      (1 MB)
  float* wsA  = bin  + (size_t)Ln * Cc * Bn * Kn;        // L*T*K*B     (64 MB)
  float* wsZ  = wsA  + (size_t)Ln * Tn * Kn * Bn;        // L*B

  float* paths = (float*)d_out;                          // [B,T,K,L]
  float* loss  = paths + (size_t)Bn * Tn * Kn * Ln;      // [L]

  ema_feat_kernel<<<Bn, 32, 0, stream>>>(lv, lvi, xf);
  chunk_op_kernel<<<dim3(Cc, Bn, Ln), 128, 0, stream>>>(xf, tr, mu, ls, logG);
  compose_fwd_kernel<<<Ln * Bn, 64, 0, stream>>>(xf, il, mu, ls, logG, ain);
  compose_bwd_kernel<<<Ln * Bn, 64, 0, stream>>>(logG, bin);
  replay_fwd_kernel<<<dim3(Cc, Ln), 128, 0, stream>>>(xf, tr, mu, ls, ain,
                                                      wsA, wsZ);
  replay_bwd_kernel<<<dim3(Cc, Ln), 128, 0, stream>>>(xf, tr, mu, ls, wsA, bin,
                                                      wsZ, paths, loss);
}
